// DeformConv2d_86328842650215
// MI455X (gfx1250) — matrix operand order repaired and checked by reading, not compiled, not run on hardware
//
#include <hip/hip_runtime.h>
#include <hip/hip_bf16.h>

typedef __attribute__((ext_vector_type(16))) __bf16 v16bf;
typedef __attribute__((ext_vector_type(8)))  float  v8f;
typedef __attribute__((ext_vector_type(4)))  int    v4i;

#define HH 64
#define WW 64
#define CIN 256
#define OUTP 256
#define NPTS 9
#define KDIM (CIN * NPTS)          // 2304
#define NPIX (4 * HH * WW)         // 16384
#define OFFCH 18
#define KSTEPS (KDIM / 32)         // 72

#if __has_builtin(__builtin_amdgcn_global_load_async_to_lds_b128) && \
    __has_builtin(__builtin_amdgcn_s_wait_asynccnt)
#define HAVE_ASYNC_LDS 1
typedef __attribute__((address_space(1))) v4i as1_v4i;
typedef __attribute__((address_space(3))) v4i as3_v4i;
#else
#define HAVE_ASYNC_LDS 0
#endif

__device__ __forceinline__ __bf16 f2bf(float f) {
    unsigned u = __builtin_bit_cast(unsigned, f);
    unsigned r = (u + 0x7FFFu + ((u >> 16) & 1u)) >> 16;   // RNE
    unsigned short s = (unsigned short)r;
    return __builtin_bit_cast(__bf16, s);
}

// ---------------- Kernel 0: w_conv fp32 -> bf16 (already K-major: o*2304 + c*9 + n)
__global__ __launch_bounds__(256) void k_cvt_w(const float* __restrict__ w, __bf16* __restrict__ a) {
    int t = blockIdx.x * 256 + threadIdx.x;      // 0 .. 589823
    a[t] = f2bf(w[t]);
}

// ---------------- Kernel 1: offset conv (fp32 direct, pad=1, stride=1)
// thread per (b, o, h, w); t = ((b*18+o)*64+h)*64+w  -> matches offset buffer layout
__global__ __launch_bounds__(256) void k_offconv(const float* __restrict__ x,
                                                 const float* __restrict__ wp,
                                                 float* __restrict__ off) {
    int t = blockIdx.x * 256 + threadIdx.x;      // 0 .. 294911
    int w = t & 63, h = (t >> 6) & 63;
    int bo = t >> 12;                            // b*18 + o
    int b = bo / OFFCH;
    int o = bo - b * OFFCH;
    const float* xb = x + (size_t)b * CIN * (HH * WW);
    float acc = 0.f;
    for (int c = 0; c < CIN; ++c) {
        const float* xc  = xb + (c << 12);
        const float* wpc = wp + ((size_t)(o * CIN + c)) * 9;
        #pragma unroll
        for (int ki = 0; ki < 3; ++ki) {
            int iy = h + ki - 1;
            #pragma unroll
            for (int kj = 0; kj < 3; ++kj) {
                int ix = w + kj - 1;
                float xv = (iy >= 0 && iy < HH && ix >= 0 && ix < WW) ? xc[(iy << 6) + ix] : 0.f;
                acc = fmaf(wpc[ki * 3 + kj], xv, acc);
            }
        }
    }
    off[t] = acc;
}

// ---------------- Kernel 2: bilinear indices + weights per (pixel, point)
__global__ __launch_bounds__(256) void k_meta(const float* __restrict__ off,
                                              int4* __restrict__ midx,
                                              float4* __restrict__ mwt) {
    int id = blockIdx.x * 256 + threadIdx.x;     // 0 .. 147455 = p*9 + n
    int n = id % NPTS;
    int p = id / NPTS;
    int b = p >> 12, hw = p & 4095;
    int h = hw >> 6, w = hw & 63;
    float ox = off[((size_t)(b * OFFCH + n) << 12) + hw];
    float oy = off[((size_t)(b * OFFCH + NPTS + n) << 12) + hw];
    float px = ox + (float)(n / 3 - 1) + (float)(h + 1);
    float py = oy + (float)(n % 3 - 1) + (float)(w + 1);
    float fx = floorf(px), fy = floorf(py);
    float ltx = fminf(fmaxf(fx,        0.f), 63.f);
    float lty = fminf(fmaxf(fy,        0.f), 63.f);
    float rbx = fminf(fmaxf(fx + 1.f,  0.f), 63.f);
    float rby = fminf(fmaxf(fy + 1.f,  0.f), 63.f);
    float pxc = fminf(fmaxf(px, 0.f), 63.f);
    float pyc = fminf(fmaxf(py, 0.f), 63.f);
    float ax = 1.f + ltx - pxc, bx = 1.f - rbx + pxc;
    float ay = 1.f + lty - pyc, by = 1.f - rby + pyc;
    int ilx = (int)ltx, ily = (int)lty, irx = (int)rbx, iry = (int)rby;
    midx[id] = make_int4(ilx * WW + ily, irx * WW + iry, ilx * WW + iry, irx * WW + ily);
    mwt[id]  = make_float4(ax * ay, bx * by, ax * by, bx * ay);
}

// ---------------- Kernel 3: build XT[p][k] bf16 (K-contiguous per pixel, coalesced writes)
__global__ __launch_bounds__(256) void k_xoff(const float* __restrict__ x,
                                              const int4* __restrict__ midx,
                                              const float4* __restrict__ mwt,
                                              __bf16* __restrict__ xt) {
    int id = blockIdx.x * 256 + threadIdx.x;     // 0 .. 37,748,735 = p*2304 + k
    int k = id % KDIM;
    int p = id / KDIM;
    int c = k / NPTS;
    int n = k - c * NPTS;
    int b = p >> 12;
    const float* xc = x + ((size_t)(b * CIN + c) << 12);
    int4   i4 = midx[p * NPTS + n];
    float4 w4 = mwt [p * NPTS + n];
    float v = w4.x * xc[i4.x] + w4.y * xc[i4.y] + w4.z * xc[i4.z] + w4.w * xc[i4.w];
    xt[id] = f2bf(v);
}

// ---------------- Kernel 4: WMMA GEMM  out[256 x 16384] = A[256 x 2304] * XT^T
// block: 256 thr (8 waves), tile 128(M) x 128(N), K-step 32, bf16 WMMA fp32-accum,
// double-buffered LDS with async cache->LDS staging when available.
__global__ __launch_bounds__(256) void k_gemm(const __bf16* __restrict__ A,
                                              const __bf16* __restrict__ Bm,
                                              float* __restrict__ out) {
    __shared__ __bf16 sA[2][128 * 40];   // padded stride 40 bf16 (80B) -> conflict-free b128
    __shared__ __bf16 sB[2][128 * 40];

    const int tid   = threadIdx.x;
    const int lane  = tid & 31;
    const int wv    = tid >> 5;
    const int mw    = wv & 1;         // 2 wave-rows  (64 M each)
    const int nw    = wv >> 1;        // 4 wave-cols  (32 N each)
    const int mTile = blockIdx.y;     // 0..1
    const int nTile = blockIdx.x;     // 0..127
    const int lane16   = lane & 15;
    const int halfLane = lane >> 4;   // 0/1

    const __bf16* gA = A  + (size_t)(mTile * 128) * KDIM;
    const __bf16* gB = Bm + (size_t)(nTile * 128) * KDIM;

    // staging decomposition: id 0..511 -> (row = id>>2, q = id&3), 16B chunks
    const int row0 = tid >> 2, q0 = tid & 3;
    const int row1 = (tid + 256) >> 2, q1 = tid & 3;

    auto stage = [&](int buf, int kk) {
#if HAVE_ASYNC_LDS
        __builtin_amdgcn_global_load_async_to_lds_b128(
            (as1_v4i*)(gA + (size_t)row0 * KDIM + kk + q0 * 8),
            (as3_v4i*)&sA[buf][row0 * 40 + q0 * 8], 0, 0);
        __builtin_amdgcn_global_load_async_to_lds_b128(
            (as1_v4i*)(gB + (size_t)row0 * KDIM + kk + q0 * 8),
            (as3_v4i*)&sB[buf][row0 * 40 + q0 * 8], 0, 0);
        __builtin_amdgcn_global_load_async_to_lds_b128(
            (as1_v4i*)(gA + (size_t)row1 * KDIM + kk + q1 * 8),
            (as3_v4i*)&sA[buf][row1 * 40 + q1 * 8], 0, 0);
        __builtin_amdgcn_global_load_async_to_lds_b128(
            (as1_v4i*)(gB + (size_t)row1 * KDIM + kk + q1 * 8),
            (as3_v4i*)&sB[buf][row1 * 40 + q1 * 8], 0, 0);
#else
        *reinterpret_cast<uint4*>(&sA[buf][row0 * 40 + q0 * 8]) =
            *reinterpret_cast<const uint4*>(&gA[(size_t)row0 * KDIM + kk + q0 * 8]);
        *reinterpret_cast<uint4*>(&sB[buf][row0 * 40 + q0 * 8]) =
            *reinterpret_cast<const uint4*>(&gB[(size_t)row0 * KDIM + kk + q0 * 8]);
        *reinterpret_cast<uint4*>(&sA[buf][row1 * 40 + q1 * 8]) =
            *reinterpret_cast<const uint4*>(&gA[(size_t)row1 * KDIM + kk + q1 * 8]);
        *reinterpret_cast<uint4*>(&sB[buf][row1 * 40 + q1 * 8]) =
            *reinterpret_cast<const uint4*>(&gB[(size_t)row1 * KDIM + kk + q1 * 8]);
#endif
    };

    v8f acc[4][2];
    #pragma unroll
    for (int mi = 0; mi < 4; ++mi)
        #pragma unroll
        for (int ni = 0; ni < 2; ++ni)
            acc[mi][ni] = (v8f){0.f,0.f,0.f,0.f,0.f,0.f,0.f,0.f};

    stage(0, 0);

    for (int i = 0; i < KSTEPS; ++i) {
        const int buf = i & 1;
#if HAVE_ASYNC_LDS
        __builtin_amdgcn_s_wait_asynccnt(0);
#endif
        __syncthreads();                       // buf ready; buf^1 free to overwrite
        if (i + 1 < KSTEPS) stage(buf ^ 1, (i + 1) * 32);

        // A fragment: lane<16 -> K{0..7,16..23}; lane>=16 -> K{8..15,24..31}
        v16bf af[4];
        #pragma unroll
        for (int mi = 0; mi < 4; ++mi) {
            int r = (mw * 64 + mi * 16 + lane16) * 40 + halfLane * 8;
            uint4* ap = reinterpret_cast<uint4*>(&af[mi]);
            ap[0] = *reinterpret_cast<const uint4*>(&sA[buf][r]);
            ap[1] = *reinterpret_cast<const uint4*>(&sA[buf][r + 16]);
        }
        // B fragment: lane<16 -> K{0..7,16..23}; lane>=16 -> K{8..15,24..31}
        v16bf bfv[2];
        #pragma unroll
        for (int ni = 0; ni < 2; ++ni) {
            int r = (nw * 32 + ni * 16 + lane16) * 40 + halfLane * 8;
            uint4* bp = reinterpret_cast<uint4*>(&bfv[ni]);
            bp[0] = *reinterpret_cast<const uint4*>(&sB[buf][r]);
            bp[1] = *reinterpret_cast<const uint4*>(&sB[buf][r + 16]);
        }
        #pragma unroll
        for (int mi = 0; mi < 4; ++mi)
            #pragma unroll
            for (int ni = 0; ni < 2; ++ni)
                acc[mi][ni] = __builtin_amdgcn_wmma_f32_16x16x32_bf16(
                    false, af[mi], false, bfv[ni], (short)0, acc[mi][ni], false, false);
    }

    // epilogue: D tile layout -> VGPR j: lanes0-15 M=j, lanes16-31 M=8+j; N=lane&15
    #pragma unroll
    for (int mi = 0; mi < 4; ++mi) {
        #pragma unroll
        for (int ni = 0; ni < 2; ++ni) {
            int m0 = mTile * 128 + mw * 64 + mi * 16 + halfLane * 8;
            int p0 = nTile * 128 + nw * 32 + ni * 16 + lane16;
            int bimg = p0 >> 12;
            int hw   = p0 & 4095;
            #pragma unroll
            for (int j = 0; j < 8; ++j)
                out[((size_t)(bimg * OUTP + m0 + j) << 12) + hw] = acc[mi][ni][j];
        }
    }
}

extern "C" void kernel_launch(void* const* d_in, const int* in_sizes, int n_in,
                              void* d_out, int out_size, void* d_ws, size_t ws_size,
                              hipStream_t stream) {
    const float* x      = (const float*)d_in[0];   // (4,256,64,64)
    const float* w_p    = (const float*)d_in[1];   // (18,256,3,3)
    const float* w_conv = (const float*)d_in[2];   // (256,256,3,3)
    float* out = (float*)d_out;                    // (4,256,64,64)

    char* ws = (char*)d_ws;
    // workspace layout (all 16B aligned)
    __bf16* Abf   = (__bf16*)(ws);                                  //  1,179,648 B
    __bf16* XT    = (__bf16*)(ws + 1179648);                        // 75,497,472 B
    float*  offb  = (float*) (ws + 1179648 + 75497472);             //  1,179,648 B
    int4*   midx  = (int4*)  (ws + 1179648 + 75497472 + 1179648);   //  2,359,296 B
    float4* mwt   = (float4*)(ws + 1179648 + 75497472 + 1179648 + 2359296);

    k_cvt_w  <<<dim3(2304),   dim3(256), 0, stream>>>(w_conv, Abf);
    k_offconv<<<dim3(1152),   dim3(256), 0, stream>>>(x, w_p, offb);
    k_meta   <<<dim3(576),    dim3(256), 0, stream>>>(offb, midx, mwt);
    k_xoff   <<<dim3(147456), dim3(256), 0, stream>>>(x, midx, mwt, XT);
    k_gemm   <<<dim3(128, 2), dim3(256), 0, stream>>>(Abf, XT, out);
}
